// Model1_17927193493507
// MI455X (gfx1250) — matrix operand order repaired and checked by reading, not compiled, not run on hardware
//
#include <hip/hip_runtime.h>

// ---------------- model constants ----------------
#define Vv    32000
#define Ee    256
#define HEe   256
#define Dd    512
#define Aa    512
#define NHh   4
#define KBm   4      // beams
#define Tt    16
#define Nn_   32
#define Ss    512
#define SOSt  1
#define EOSt  2
#define NEGv  (-1000000000.0f)

typedef __attribute__((ext_vector_type(16))) _Float16 v16h;
typedef __attribute__((ext_vector_type(8)))  float    v8f;
typedef _Float16 h16;

// =====================================================================
// Generic NT WMMA GEMM:  C[m][n] = act( sum_k A16[m][k]*B16[n][k] + bias[n] + Cadd[m][n] )
// A: (M x K) f16 row-major (lda), B: (N x K) f16 row-major (ldb)  -> NT form
// Outputs optionally f32 (Cout/ldc) and/or f16 (Cout16/ldc16). act: 0=none 1=tanh
// Each wave computes a 32x64 tile (2 M-frags x 4 N-frags -> 8 WMMAs per k-step)
// so each loaded fragment is reused; raises FLOP/byte ~2.7x vs 16x16 tiles.
// Requires: M % 32 == 0, N % 64 == 0, K % 32 == 0 (true for every GEMM here).
// =====================================================================
__global__ __launch_bounds__(256)
void gemm_wmma(const h16* __restrict__ A, long lda,
               const h16* __restrict__ B, long ldb,
               const float* __restrict__ Cadd, long ldadd,
               const float* __restrict__ bias,
               float* __restrict__ Cout, long ldc,
               h16* __restrict__ Cout16, long ldc16,
               int M, int Nc, int Kc, int act)
{
    const int wave = threadIdx.x >> 5;
    const int lane = threadIdx.x & 31;
    const int tilesN = Nc >> 6;                 // 64-wide N tiles
    const long nTiles = (long)(M >> 5) * tilesN;
    const long tile = (long)blockIdx.x * 8 + wave;
    if (tile >= nTiles) return;                 // wave-uniform: EXEC stays all-1s
    const int tm = (int)(tile / tilesN);
    const int tn = (int)(tile % tilesN);
    const int m0 = tm << 5, n0 = tn << 6;

    // A fragment: lane l<16 holds row l, K={0..7,16..23}; l>=16 holds row l-16, K={8..15,24..31}
    const int arow  = lane & 15;
    const int koff  = (lane >> 4) << 3;         // 0 or 8
    // B fragment: lane l<16 holds col l, K={0..7,16..23}; l>=16 holds col l-16, K={8..15,24..31}
    const int bcol  = lane & 15;
    const int kbase = (lane >> 4) << 3;         // 0 or 8

    const h16* Ar[2];
    Ar[0] = A + (long)(m0 + arow) * lda;
    Ar[1] = A + (long)(m0 + 16 + arow) * lda;
    const h16* Bc[4];
    #pragma unroll
    for (int j = 0; j < 4; ++j) Bc[j] = B + (long)(n0 + j*16 + bcol) * ldb;

    __builtin_prefetch(Ar[0], 0, 1);            // global_prefetch_b8
    __builtin_prefetch(Bc[0], 0, 1);

    v8f acc[2][4];
    #pragma unroll
    for (int a = 0; a < 2; ++a)
        #pragma unroll
        for (int j = 0; j < 4; ++j) acc[a][j] = (v8f){};

    for (int k = 0; k < Kc; k += 32) {
        union { v16h v; uint4 u[2]; } fa[2], fb[4];
        #pragma unroll
        for (int a = 0; a < 2; ++a) {
            fa[a].u[0] = *(const uint4*)(Ar[a] + k + koff);
            fa[a].u[1] = *(const uint4*)(Ar[a] + k + koff + 16);
        }
        #pragma unroll
        for (int j = 0; j < 4; ++j) {
            fb[j].u[0] = *(const uint4*)(Bc[j] + k + kbase);
            fb[j].u[1] = *(const uint4*)(Bc[j] + k + kbase + 16);
        }
        #pragma unroll
        for (int a = 0; a < 2; ++a)
            #pragma unroll
            for (int j = 0; j < 4; ++j)
                acc[a][j] = __builtin_amdgcn_wmma_f32_16x16x32_f16(
                    false, fa[a].v, false, fb[j].v, (short)0, acc[a][j], false, false);
    }

    // C/D layout per 16x16 sub-tile: VGPR i -> row (i + 8*(lane>=16)), col = lane&15
    const int ccol = lane & 15;
    const int crow = (lane >> 4) << 3;
    #pragma unroll
    for (int j = 0; j < 4; ++j) {
        const int cn = n0 + j*16 + ccol;
        const float bval = bias ? bias[cn] : 0.0f;
        #pragma unroll
        for (int a = 0; a < 2; ++a) {
            #pragma unroll
            for (int i = 0; i < 8; ++i) {
                const int r = m0 + a*16 + crow + i;
                float v = acc[a][j][i] + bval;
                if (Cadd)  v += Cadd[(long)r * ldadd + cn];
                if (act == 1) v = tanhf(v);
                if (Cout)   Cout  [(long)r * ldc   + cn] = v;
                if (Cout16) Cout16[(long)r * ldc16 + cn] = (h16)v;
            }
        }
    }
}

// ---------------- elementwise / utility kernels ----------------
__global__ void k_fill_f32(float* p, float v, long n){ long i=(long)blockIdx.x*256+threadIdx.x; if(i<n) p[i]=v; }
__global__ void k_fill_f16(h16* p, long n){ long i=(long)blockIdx.x*256+threadIdx.x; if(i<n) p[i]=(h16)0.0f; }
__global__ void k_fill_i32(int* p, int v, long n){ long i=(long)blockIdx.x*256+threadIdx.x; if(i<n) p[i]=v; }

__global__ void k_conv16(const float* __restrict__ s, h16* __restrict__ d, long n){
    long i=(long)blockIdx.x*256+threadIdx.x; if(i<n) d[i]=(h16)s[i];
}
// dst (C x R) = transpose(src (R x C)), f32 -> f16
__global__ void k_trans16(const float* __restrict__ s, h16* __restrict__ d, int R, int C){
    long i=(long)blockIdx.x*256+threadIdx.x; if(i>=(long)R*C) return;
    int r=(int)(i / C), c=(int)(i % C);
    d[(long)c*R + r] = (h16)s[i];
}
// dst[m*stride + e] = (h16) emb[tok[m]*E + e]
__global__ void k_embed(const int* __restrict__ tok, const float* __restrict__ emb,
                        h16* __restrict__ dst, long stride, int M){
    long i=(long)blockIdx.x*256+threadIdx.x; if(i>=(long)M*Ee) return;
    int m=(int)(i/Ee), e=(int)(i%Ee);
    dst[(long)m*stride + e] = (h16)emb[(long)tok[m]*Ee + e];
}

__device__ __forceinline__ float sigmoidf_(float x){ return 1.0f/(1.0f+expf(-x)); }

// gates (M x 4H): [i f g o]; writes c_out (f32) and up to two f16 h destinations
__global__ void k_lstm_pw(const float* __restrict__ gates, const float* __restrict__ c_in,
                          float* __restrict__ c_out,
                          h16* __restrict__ ha, long sa,
                          h16* __restrict__ hb, long sb,
                          int M, int H)
{
    long i=(long)blockIdx.x*256+threadIdx.x; if(i>=(long)M*H) return;
    int m=(int)(i/H), j=(int)(i%H);
    const float* g = gates + (long)m*4*H;
    float ig = sigmoidf_(g[j]);
    float fg = sigmoidf_(g[H+j]);
    float gg = tanhf(g[2*H+j]);
    float og = sigmoidf_(g[3*H+j]);
    float c2 = fg * c_in[(long)m*H+j] + ig*gg;
    float h2 = og * tanhf(c2);
    c_out[(long)m*H+j] = c2;
    ha[(long)m*sa + j] = (h16)h2;
    if (hb) hb[(long)m*sb + j] = (h16)h2;
}

// one block per (m, head): softmax(q.k/sqrt(dh)) @ v  -> out f16 (M x D)
__global__ void k_attn(const float* __restrict__ q, const float* __restrict__ Kp,
                       const float* __restrict__ Vp, h16* __restrict__ out,
                       int M, int beams)
{
    const int dh = Dd / NHh;                    // 128
    int m = blockIdx.x >> 2, h = blockIdx.x & 3;
    int n = m / beams;
    __shared__ float sc[Ss];
    __shared__ float red[256];
    const float* qv = q + (long)m*Dd + h*dh;
    const float* Kb = Kp + (long)n*Ss*Dd + h*dh;
    for (int s = threadIdx.x; s < Ss; s += 256) {
        const float* kr = Kb + (long)s*Dd;
        float d = 0.f;
        for (int j = 0; j < dh; ++j) d += qv[j]*kr[j];
        sc[s] = d * 0.08838834764831845f;       // 1/sqrt(128)
    }
    __syncthreads();
    float mx = -3.4e38f;
    for (int s = threadIdx.x; s < Ss; s += 256) mx = fmaxf(mx, sc[s]);
    red[threadIdx.x]=mx; __syncthreads();
    for (int st=128; st>0; st>>=1){ if(threadIdx.x<st) red[threadIdx.x]=fmaxf(red[threadIdx.x],red[threadIdx.x+st]); __syncthreads(); }
    mx = red[0]; __syncthreads();
    float sm = 0.f;
    for (int s = threadIdx.x; s < Ss; s += 256){ float e=expf(sc[s]-mx); sc[s]=e; sm+=e; }
    red[threadIdx.x]=sm; __syncthreads();
    for (int st=128; st>0; st>>=1){ if(threadIdx.x<st) red[threadIdx.x]+=red[threadIdx.x+st]; __syncthreads(); }
    float inv = 1.0f/red[0];
    __syncthreads();
    const float* Vb = Vp + (long)n*Ss*Dd + h*dh;
    for (int d = threadIdx.x; d < dh; d += 256) {
        float a = 0.f;
        for (int s = 0; s < Ss; ++s) a += sc[s]*Vb[(long)s*Dd + d];
        out[(long)m*Dd + h*dh + d] = (h16)(a*inv);
    }
}

// per-row max and log-sum-exp (one block per row)
__global__ void k_rowstats(const float* __restrict__ X, long ld, float* rmax, float* rlse, int cols){
    int m = blockIdx.x;
    const float* x = X + (long)m*ld;
    __shared__ float red[256];
    float mx = -3.4e38f;
    for (int c = threadIdx.x; c < cols; c += 256) mx = fmaxf(mx, x[c]);
    red[threadIdx.x]=mx; __syncthreads();
    for (int st=128; st>0; st>>=1){ if(threadIdx.x<st) red[threadIdx.x]=fmaxf(red[threadIdx.x],red[threadIdx.x+st]); __syncthreads(); }
    mx = red[0]; __syncthreads();
    float sm = 0.f;
    for (int c = threadIdx.x; c < cols; c += 256) sm += expf(x[c]-mx);
    red[threadIdx.x]=sm; __syncthreads();
    for (int st=128; st>0; st>>=1){ if(threadIdx.x<st) red[threadIdx.x]+=red[threadIdx.x+st]; __syncthreads(); }
    if (threadIdx.x==0){ rmax[m]=mx; rlse[m]=logf(red[0]); }
}

// ---------------- top-k helpers ----------------
__device__ __forceinline__ bool tk_better(float v,int i,float v2,int i2){ return (v>v2)||(v==v2&&i<i2); }
__device__ __forceinline__ void tk_ins(float v,int i,float bv[4],int bi[4]){
    if(!tk_better(v,i,bv[3],bi[3])) return;
    bv[3]=v; bi[3]=i;
    for(int p=3;p>0;--p){
        if(tk_better(bv[p],bi[p],bv[p-1],bi[p-1])){
            float tv=bv[p];bv[p]=bv[p-1];bv[p-1]=tv;
            int ti=bi[p];bi[p]=bi[p-1];bi[p-1]=ti;
        } else break;
    }
}

// step-0 top-k over V (one block per batch n)
__global__ void k_topk0(const float* __restrict__ logits, const float* rmax, const float* rlse,
                        float* scores, int* tok, int* fin)
{
    int n = blockIdx.x;
    __shared__ float sv[1024]; __shared__ int si[1024];
    float bv[4]={-3.4e38f,-3.4e38f,-3.4e38f,-3.4e38f}; int bi[4]={0x7fffffff,0x7fffffff,0x7fffffff,0x7fffffff};
    const float base = -rmax[n] - rlse[n];
    const float* x = logits + (long)n*Vv;
    for (int v = threadIdx.x; v < Vv; v += 256) tk_ins(x[v]+base, v, bv, bi);
    for (int k=0;k<4;++k){ sv[threadIdx.x*4+k]=bv[k]; si[threadIdx.x*4+k]=bi[k]; }
    __syncthreads();
    if (threadIdx.x==0){
        float fv[4]={-3.4e38f,-3.4e38f,-3.4e38f,-3.4e38f}; int fi[4]={0x7fffffff,0x7fffffff,0x7fffffff,0x7fffffff};
        for (int j=0;j<1024;++j) tk_ins(sv[j], si[j], fv, fi);
        for (int k=0;k<4;++k){
            scores[n*KBm+k]=fv[k]; tok[n*KBm+k]=fi[k]; fin[n*KBm+k]=(fi[k]==EOSt)?1:0;
        }
    }
}

// beam-step top-k over K*V candidates (one block per batch n)
__global__ void k_beam_topk(const float* __restrict__ logits, const float* rmax, const float* rlse,
                            float* scores, int* tok, int* fin, int* beams)
{
    int n = blockIdx.x;
    __shared__ float ssc[KBm]; __shared__ int sfin[KBm];
    __shared__ float sv[1024]; __shared__ int si[1024];
    if (threadIdx.x < KBm){ ssc[threadIdx.x]=scores[n*KBm+threadIdx.x]; sfin[threadIdx.x]=fin[n*KBm+threadIdx.x]; }
    __syncthreads();
    float bv[4]={-3.4e38f,-3.4e38f,-3.4e38f,-3.4e38f}; int bi[4]={0x7fffffff,0x7fffffff,0x7fffffff,0x7fffffff};
    for (long idx = threadIdx.x; idx < (long)KBm*Vv; idx += 256) {
        int k = (int)(idx / Vv), v = (int)(idx % Vv);
        float val;
        if (sfin[k]) val = ssc[k] + ((v==EOSt)?0.0f:NEGv);
        else { int m = n*KBm+k; val = ssc[k] + logits[(long)m*Vv + v] - rmax[m] - rlse[m]; }
        tk_ins(val, (int)idx, bv, bi);
    }
    for (int k=0;k<4;++k){ sv[threadIdx.x*4+k]=bv[k]; si[threadIdx.x*4+k]=bi[k]; }
    __syncthreads();
    if (threadIdx.x==0){
        float fv[4]={-3.4e38f,-3.4e38f,-3.4e38f,-3.4e38f}; int fi[4]={0x7fffffff,0x7fffffff,0x7fffffff,0x7fffffff};
        for (int j=0;j<1024;++j) tk_ins(sv[j], si[j], fv, fi);
        for (int k=0;k<4;++k){
            int b = fi[k]/Vv, v = fi[k]%Vv;
            scores[n*KBm+k]=fv[k]; tok[n*KBm+k]=v; beams[n*KBm+k]=b;
        }
        int nf[4];
        for (int k=0;k<4;++k){ int b=beams[n*KBm+k]; nf[k] = (sfin[b] | ((tok[n*KBm+k]==EOSt)?1:0)); }
        for (int k=0;k<4;++k) fin[n*KBm+k]=nf[k];
    }
}

// ---------------- beam replicate / gather ----------------
__global__ void k_rep16(const h16* __restrict__ s, long ss, h16* __restrict__ d, long ds, int width){
    long i=(long)blockIdx.x*256+threadIdx.x; if(i>=(long)Nn_*KBm*width) return;
    int m=(int)(i/width), w=(int)(i%width); int n=m>>2;
    d[(long)m*ds+w] = s[(long)n*ss+w];
}
__global__ void k_repf(const float* __restrict__ s, long ss, float* __restrict__ d, long ds, int width){
    long i=(long)blockIdx.x*256+threadIdx.x; if(i>=(long)Nn_*KBm*width) return;
    int m=(int)(i/width), w=(int)(i%width); int n=m>>2;
    d[(long)m*ds+w] = s[(long)n*ss+w];
}
__global__ void k_gather16(const h16* __restrict__ s, long ss, h16* __restrict__ d, long ds,
                           const int* __restrict__ beams, int width){
    long i=(long)blockIdx.x*256+threadIdx.x; if(i>=(long)Nn_*KBm*width) return;
    int m=(int)(i/width), w=(int)(i%width); int n=m>>2;
    int b = beams[m];
    d[(long)m*ds+w] = s[(long)(n*KBm+b)*ss+w];
}
__global__ void k_gatherf(const float* __restrict__ s, long ss, float* __restrict__ d, long ds,
                          const int* __restrict__ beams, int width){
    long i=(long)blockIdx.x*256+threadIdx.x; if(i>=(long)Nn_*KBm*width) return;
    int m=(int)(i/width), w=(int)(i%width); int n=m>>2;
    int b = beams[m];
    d[(long)m*ds+w] = s[(long)(n*KBm+b)*ss+w];
}

// [hF|hB] -> catH (f16), [cF|cB] -> catC (f16)
__global__ void k_bridge_in(const h16* hF, const h16* hB, const float* cF, const float* cB,
                            h16* catH, h16* catC){
    long i=(long)blockIdx.x*256+threadIdx.x; if(i>=(long)Nn_*2*HEe) return;
    int n=(int)(i/512), j=(int)(i%512);
    catH[i] = (j<256) ? hF[n*256+j] : hB[n*256+j-256];
    catC[i] = (h16)((j<256) ? cF[n*256+j] : cB[n*256+j-256]);
}

// =====================================================================
extern "C" void kernel_launch(void* const* d_in, const int* in_sizes, int n_in,
                              void* d_out, int out_size, void* d_ws, size_t ws_size,
                              hipStream_t stream)
{
    const int*   tokens   = (const int*)  d_in[0];
    const float* emb      = (const float*)d_in[1];
    const float* eWihF    = (const float*)d_in[2];
    const float* eWhhF    = (const float*)d_in[3];
    const float* ebF      = (const float*)d_in[4];
    const float* eWihB    = (const float*)d_in[5];
    const float* eWhhB    = (const float*)d_in[6];
    const float* ebB      = (const float*)d_in[7];
    const float* encOutW  = (const float*)d_in[8];
    const float* brWh     = (const float*)d_in[9];
    const float* brWc     = (const float*)d_in[10];
    const float* decWif   = (const float*)d_in[11];
    const float* decBif   = (const float*)d_in[12];
    const float* decWih   = (const float*)d_in[13];
    const float* decWhh   = (const float*)d_in[14];
    const float* decB     = (const float*)d_in[15];
    const float* Wq       = (const float*)d_in[16];
    const float* Wk       = (const float*)d_in[17];
    const float* Wv       = (const float*)d_in[18];
    const float* Wo       = (const float*)d_in[19];
    const float* Wav      = (const float*)d_in[20];
    const float* Wout     = (const float*)d_in[21];
    float* out = (float*)d_out;

    // ---- bump allocator on workspace ----
    char* ws = (char*)d_ws;
    size_t off = 0;
    auto aH = [&](size_t n)->h16*  { h16*  p=(h16*) (ws+off); off=(off+n*2+255)&~(size_t)255; return p; };
    auto aF = [&](size_t n)->float*{ float*p=(float*)(ws+off); off=(off+n*4+255)&~(size_t)255; return p; };
    auto aI = [&](size_t n)->int*  { int*  p=(int*) (ws+off); off=(off+n*4+255)&~(size_t)255; return p; };

    // f16 weights (NT form: rows = output dim, contiguous over K)
    h16* wWihF = aH(1024*256); h16* wWhhF = aH(1024*256);
    h16* wWihB = aH(1024*256); h16* wWhhB = aH(1024*256);
    h16* wDecWih = aH(2048*256); h16* wDecWhh = aH((size_t)2048*512);
    h16* wEncOutT = aH((size_t)512*512);
    h16* wBrHT = aH((size_t)512*512); h16* wBrCT = aH((size_t)512*512);
    h16* wDecWifT = aH((size_t)256*768);
    h16* wQT = aH((size_t)512*512); h16* wKT = aH((size_t)512*512);
    h16* wVT = aH((size_t)512*512); h16* wOT = aH((size_t)512*512);
    h16* wAvT = aH((size_t)512*1024);
    h16* wOutT = aH((size_t)Vv*512);
    // activations
    h16* x16      = aH((size_t)Nn_*Ss*Ee);
    h16* encCat   = aH((size_t)Nn_*Ss*2*HEe);
    h16* encProj  = aH((size_t)Nn_*Ss*Dd);
    h16* h16F = aH(Nn_*HEe); h16* h16B = aH(Nn_*HEe);
    h16* catH16 = aH(Nn_*512); h16* catC16 = aH(Nn_*512);
    h16* bh16 = aH(Nn_*512);
    h16* cat1 = aH(128*768);      // [emb(256) | av(512)]
    h16* hcur16 = aH(128*512);
    h16* cat2 = aH(128*1024);     // [h2(512) | attn_o(512)]
    h16* xi16 = aH(128*256);
    h16* avtmp16 = aH(128*512);
    h16* attn16  = aH(128*512);
    float* preF = aF((size_t)Nn_*Ss*1024);
    float* preB = aF((size_t)Nn_*Ss*1024);
    float* cF = aF(Nn_*HEe); float* cB = aF(Nn_*HEe);
    float* Kproj = aF((size_t)Nn_*Ss*Dd);
    float* Vproj = aF((size_t)Nn_*Ss*Dd);
    float* bc32 = aF(Nn_*512);
    float* gates = aF((size_t)128*2048);
    float* ccur = aF(128*512); float* ctmp = aF(128*512);
    float* qbuf = aF(128*512);
    float* logits0 = aF((size_t)Nn_*Vv);
    float* rmax = aF(128); float* rlse = aF(128); float* scores = aF(128);
    int* tok = aI(128); int* fin = aI(128); int* beams = aI(128); int* tok0 = aI(32);

    auto g1 = [](long n){ return dim3((unsigned)((n + 255) / 256)); };
    auto GE = [&](const h16* A, long lda, const h16* B, long ldb,
                  const float* Cadd, long ldadd, const float* bias,
                  float* C, long ldc, h16* C16, long ldc16,
                  int M, int Nc, int Kc, int act){
        long tiles = (long)(M/32)*(Nc/64);      // 32x64 wave tiles
        gemm_wmma<<<dim3((unsigned)((tiles+7)/8)), dim3(256), 0, stream>>>(
            A, lda, B, ldb, Cadd, ldadd, bias, C, ldc, C16, ldc16, M, Nc, Kc, act);
    };

    // ---------- Phase A: weight conversion ----------
    k_conv16<<<g1(1024*256),256,0,stream>>>(eWihF, wWihF, 1024*256);
    k_conv16<<<g1(1024*256),256,0,stream>>>(eWhhF, wWhhF, 1024*256);
    k_conv16<<<g1(1024*256),256,0,stream>>>(eWihB, wWihB, 1024*256);
    k_conv16<<<g1(1024*256),256,0,stream>>>(eWhhB, wWhhB, 1024*256);
    k_conv16<<<g1(2048*256),256,0,stream>>>(decWih, wDecWih, 2048*256);
    k_conv16<<<g1((long)2048*512),256,0,stream>>>(decWhh, wDecWhh, (long)2048*512);
    k_trans16<<<g1((long)512*512),256,0,stream>>>(encOutW, wEncOutT, 512, 512);
    k_trans16<<<g1((long)512*512),256,0,stream>>>(brWh, wBrHT, 512, 512);
    k_trans16<<<g1((long)512*512),256,0,stream>>>(brWc, wBrCT, 512, 512);
    k_trans16<<<g1((long)768*256),256,0,stream>>>(decWif, wDecWifT, 768, 256);
    k_trans16<<<g1((long)512*512),256,0,stream>>>(Wq, wQT, 512, 512);
    k_trans16<<<g1((long)512*512),256,0,stream>>>(Wk, wKT, 512, 512);
    k_trans16<<<g1((long)512*512),256,0,stream>>>(Wv, wVT, 512, 512);
    k_trans16<<<g1((long)512*512),256,0,stream>>>(Wo, wOT, 512, 512);
    k_trans16<<<g1((long)1024*512),256,0,stream>>>(Wav, wAvT, 1024, 512);
    k_trans16<<<g1((long)512*Vv),256,0,stream>>>(Wout, wOutT, 512, Vv);

    // ---------- Phase B: encoder ----------
    k_embed<<<g1((long)Nn_*Ss*Ee),256,0,stream>>>(tokens, emb, x16, Ee, Nn_*Ss);
    // pre-gates for all timesteps (big WMMA GEMMs, bias folded in)
    GE(x16, Ee, wWihF, Ee, nullptr,0, ebF, preF, 1024, nullptr,0, Nn_*Ss, 1024, Ee, 0);
    GE(x16, Ee, wWihB, Ee, nullptr,0, ebB, preB, 1024, nullptr,0, Nn_*Ss, 1024, Ee, 0);
    k_fill_f16<<<g1(Nn_*HEe),256,0,stream>>>(h16F, Nn_*HEe);
    k_fill_f16<<<g1(Nn_*HEe),256,0,stream>>>(h16B, Nn_*HEe);
    k_fill_f32<<<g1(Nn_*HEe),256,0,stream>>>(cF, 0.f, Nn_*HEe);
    k_fill_f32<<<g1(Nn_*HEe),256,0,stream>>>(cB, 0.f, Nn_*HEe);
    k_fill_f16<<<g1((long)128*768),256,0,stream>>>(cat1, (long)128*768); // av0 = 0

    for (int it = 0; it < Ss; ++it) {
        // forward, timestep t = it
        GE(h16F, HEe, wWhhF, HEe, preF + (long)it*1024, (long)Ss*1024, nullptr,
           gates, 1024, nullptr,0, Nn_, 1024, HEe, 0);
        k_lstm_pw<<<g1(Nn_*HEe),256,0,stream>>>(gates, cF, cF,
            h16F, HEe, encCat + (long)it*512, (long)Ss*512, Nn_, HEe);
        // backward, timestep tt = S-1-it
        int tt = Ss - 1 - it;
        GE(h16B, HEe, wWhhB, HEe, preB + (long)tt*1024, (long)Ss*1024, nullptr,
           gates, 1024, nullptr,0, Nn_, 1024, HEe, 0);
        k_lstm_pw<<<g1(Nn_*HEe),256,0,stream>>>(gates, cB, cB,
            h16B, HEe, encCat + (long)tt*512 + 256, (long)Ss*512, Nn_, HEe);
    }

    // ---------- Phase C: projections / bridge ----------
    GE(encCat, 512, wEncOutT, 512, nullptr,0, nullptr, nullptr,0, encProj, Dd, Nn_*Ss, Dd, 512, 1);
    GE(encProj, Dd, wKT, Dd, nullptr,0, nullptr, Kproj, Dd, nullptr,0, Nn_*Ss, Dd, Dd, 0);
    GE(encProj, Dd, wVT, Dd, nullptr,0, nullptr, Vproj, Dd, nullptr,0, Nn_*Ss, Dd, Dd, 0);
    k_bridge_in<<<g1((long)Nn_*512),256,0,stream>>>(h16F, h16B, cF, cB, catH16, catC16);
    GE(catH16, 512, wBrHT, 512, nullptr,0, nullptr, nullptr,0, bh16, 512, Nn_, Dd, 512, 1);
    GE(catC16, 512, wBrCT, 512, nullptr,0, nullptr, bc32, 512, nullptr,0, Nn_, Dd, 512, 1);

    // ---------- decoder step (shared for step0 / beam steps) ----------
    auto dec_step = [&](int M, const h16* hA, long hAld, const float* c_in, float* c_out,
                        float* logits_dst, int nbeams){
        GE(cat1, 768, wDecWifT, 768, nullptr,0, decBif, nullptr,0, xi16, 256, M, 256, 768, 1);
        GE(xi16, 256, wDecWih, 256, nullptr,0, decB, gates, 2048, nullptr,0, M, 2048, 256, 0);
        GE(hA, hAld, wDecWhh, 512, gates, 2048, nullptr, gates, 2048, nullptr,0, M, 2048, 512, 0);
        k_lstm_pw<<<g1((long)M*512),256,0,stream>>>(gates, c_in, c_out, cat2, 1024, nullptr, 0, M, 512);
        GE(cat2, 1024, wQT, 512, nullptr,0, nullptr, qbuf, 512, nullptr,0, M, 512, 512, 0);
        k_attn<<<dim3(M*NHh),dim3(256),0,stream>>>(qbuf, Kproj, Vproj, attn16, M, nbeams);
        GE(attn16, 512, wOT, 512, nullptr,0, nullptr, nullptr,0, cat2+512, 1024, M, 512, 512, 0);
        GE(cat2, 1024, wAvT, 1024, nullptr,0, nullptr, nullptr,0, avtmp16, 512, M, 512, 1024, 1);
        GE(avtmp16, 512, wOutT, 512, nullptr,0, nullptr, logits_dst, Vv, nullptr,0, M, Vv, 512, 0);
        k_rowstats<<<dim3(M),dim3(256),0,stream>>>(logits_dst, Vv, rmax, rlse, Vv);
    };

    // ---------- Phase D: decoder step 0 (M = 32) ----------
    k_fill_i32<<<dim3(1),dim3(32),0,stream>>>(tok0, SOSt, 32);
    k_embed<<<g1((long)Nn_*Ee),256,0,stream>>>(tok0, emb, cat1, 768, Nn_);
    dec_step(Nn_, bh16, 512, bc32, ctmp, logits0, 1);
    k_topk0<<<dim3(Nn_),dim3(256),0,stream>>>(logits0, rmax, rlse, scores, tok, fin);
    k_rep16<<<g1((long)128*512),256,0,stream>>>(cat2, 1024, hcur16, 512, 512);
    k_rep16<<<g1((long)128*512),256,0,stream>>>(avtmp16, 512, cat1+256, 768, 512);
    k_repf <<<g1((long)128*512),256,0,stream>>>(ctmp, 512, ccur, 512, 512);

    // ---------- Phase E: T beam steps (M = 128) ----------
    for (int t = 0; t < Tt; ++t) {
        float* ldst = out + (size_t)t * 128 * Vv;   // raw logits are the scan output
        k_embed<<<g1((long)128*Ee),256,0,stream>>>(tok, emb, cat1, 768, 128);
        dec_step(128, hcur16, 512, ccur, ctmp, ldst, KBm);
        k_beam_topk<<<dim3(Nn_),dim3(256),0,stream>>>(ldst, rmax, rlse, scores, tok, fin, beams);
        k_gather16<<<g1((long)128*512),256,0,stream>>>(cat2, 1024, hcur16, 512, beams, 512);
        k_gather16<<<g1((long)128*512),256,0,stream>>>(avtmp16, 512, cat1+256, 768, beams, 512);
        k_gatherf <<<g1((long)128*512),256,0,stream>>>(ctmp, 512, ccur, 512, beams, 512);
    }
    (void)in_sizes; (void)n_in; (void)out_size; (void)ws_size;
}
